// ExtraMSAStackBlock_2697239462210
// MI455X (gfx1250) — matrix operand order repaired and checked by reading, not compiled, not run on hardware
//
#include <hip/hip_runtime.h>

// ---------------------------------------------------------------------------
// Types for CDNA5 WMMA (wave32, v_wmma_f32_16x16x32_bf16)
// ---------------------------------------------------------------------------
typedef __attribute__((ext_vector_type(16))) __bf16 v16bf;
typedef __attribute__((ext_vector_type(8)))  __bf16 v8bf;
typedef __attribute__((ext_vector_type(8)))  float  v8f;
typedef unsigned short u16b;

__device__ __forceinline__ u16b f2bf(float x) {
  unsigned u = __float_as_uint(x);
  u += 0x7FFFu + ((u >> 16) & 1u);   // round-to-nearest-even
  return (u16b)(u >> 16);
}

// ---------------------------------------------------------------------------
// Generic batched WMMA GEMM:  C = act(scale * A @ B^T + bias)
//   A  : bf16 [batch][M][K]   row-major, lda = row stride (elems), sA batch stride
//   Bt : bf16 [batch][N][K]   (i.e. B column-major), ldb row stride, sB batch stride
//   C  : f32 with configurable strides (opst==0), or
//        opst==1: bf16 store permuted for outer-product-mean:
//                 (row=i*32+a, col=j*32+b) -> dst[(i*256+j)*1024 + a*32 + b]
// One wave computes a 32x32 macro-tile (2x2 WMMA tiles, fragment reuse).
// 8 waves per block. Requires M%32==0, N%32==0, K%32==0, lda/ldb%8==0.
// ---------------------------------------------------------------------------
__global__ __launch_bounds__(256)
void wmma_gemm_bf16_k(const __bf16* __restrict__ A, long lda, long sA,
                      const __bf16* __restrict__ Bt, long ldb, long sB,
                      void* __restrict__ Cp, long ldcM, long ldcN, long sC,
                      const float* __restrict__ bias, float scale,
                      int N, int K, int tilesN, int tilesTotal, int act, int opst)
{
  (void)N;
  int wave = threadIdx.x >> 5;
  int tile = blockIdx.x * 8 + wave;
  if (tile >= tilesTotal) return;          // uniform per-wave guard
  int tn = tile % tilesN;                  // tilesN = N/32
  int tm = tile / tilesN;
  int b  = blockIdx.y;
  int lane = threadIdx.x & 31;
  int l16 = lane & 15, lh = lane >> 4;

  // A fragment layout: lane holds row M=l16; e<8 -> K=lh*8+e ; e>=8 -> K=16+lh*8+(e-8)
  const __bf16* arow[2];
  arow[0] = A + (long)b * sA + ((long)tm * 32 + l16) * lda + lh * 8;
  arow[1] = arow[0] + 16 * lda;
  // B fragment layout: lane holds col N=l16; e<8 -> K=lh*8+e ; e>=8 -> K=16+lh*8+(e-8)
  const __bf16* brow[2];
  brow[0] = Bt + (long)b * sB + ((long)tn * 32 + l16) * ldb + lh * 8;
  brow[1] = brow[0] + 16 * ldb;

  v8f acc[2][2] = {};
#pragma unroll 2
  for (int kk = 0; kk < K; kk += 32) {
    if (kk + 512 < K) {                    // uniform: prefetch next K-panel (global_prefetch_b8)
      __builtin_prefetch(arow[0] + kk + 512, 0, 1);
      __builtin_prefetch(brow[0] + kk + 512, 0, 1);
    }
    v16bf af[2], bfm[2];
#pragma unroll
    for (int i = 0; i < 2; ++i) {
      v8bf a0 = *(const v8bf*)(arow[i] + kk);
      v8bf a1 = *(const v8bf*)(arow[i] + kk + 16);
      v8bf b0 = *(const v8bf*)(brow[i] + kk);
      v8bf b1 = *(const v8bf*)(brow[i] + kk + 16);
#pragma unroll
      for (int e = 0; e < 8; ++e) {
        af[i][e] = a0[e]; af[i][e + 8] = a1[e];
        bfm[i][e] = b0[e]; bfm[i][e + 8] = b1[e];
      }
    }
#pragma unroll
    for (int i = 0; i < 2; ++i)
#pragma unroll
      for (int j = 0; j < 2; ++j)
        acc[i][j] = __builtin_amdgcn_wmma_f32_16x16x32_bf16(false, af[i], false, bfm[j],
                                                            (short)0, acc[i][j], false, false);
  }

  // ---- epilogue: scale + bias (uniform passes, no per-element branches) ----
#pragma unroll
  for (int tj = 0; tj < 2; ++tj) {
    float bv = bias ? bias[tn * 32 + tj * 16 + l16] : 0.0f;
#pragma unroll
    for (int ti = 0; ti < 2; ++ti)
#pragma unroll
      for (int r = 0; r < 8; ++r)
        acc[ti][tj][r] = acc[ti][tj][r] * scale + bv;
  }
  if (act == 1) {
#pragma unroll
    for (int tj = 0; tj < 2; ++tj)
#pragma unroll
      for (int ti = 0; ti < 2; ++ti)
#pragma unroll
        for (int r = 0; r < 8; ++r)
          acc[ti][tj][r] = fmaxf(acc[ti][tj][r], 0.0f);
  } else if (act == 2) {
#pragma unroll
    for (int tj = 0; tj < 2; ++tj)
#pragma unroll
      for (int ti = 0; ti < 2; ++ti)
#pragma unroll
        for (int r = 0; r < 8; ++r)
          acc[ti][tj][r] = __fdividef(1.0f, 1.0f + __expf(-acc[ti][tj][r]));
  }
  if (opst == 0) {
    float* Cf = (float*)Cp + (long)b * sC;
#pragma unroll
    for (int tj = 0; tj < 2; ++tj) {
      int col = tn * 32 + tj * 16 + l16;
#pragma unroll
      for (int ti = 0; ti < 2; ++ti)
#pragma unroll
        for (int r = 0; r < 8; ++r) {
          int row = tm * 32 + ti * 16 + lh * 8 + r;   // D layout: vgpr r -> M = r + 8*(lane/16)
          Cf[(long)row * ldcM + (long)col * ldcN] = acc[ti][tj][r];
        }
    }
  } else {
    u16b* Ch = (u16b*)Cp;
#pragma unroll
    for (int tj = 0; tj < 2; ++tj) {
      int col = tn * 32 + tj * 16 + l16;
#pragma unroll
      for (int ti = 0; ti < 2; ++ti)
#pragma unroll
        for (int r = 0; r < 8; ++r) {
          int row = tm * 32 + ti * 16 + lh * 8 + r;
          long idx = ((long)(row >> 5) * 256 + (col >> 5)) * 1024 + ((row & 31) << 5) + (col & 31);
          Ch[idx] = f2bf(acc[ti][tj][r]);
        }
    }
  }
}

// ---------------------------------------------------------------------------
// LayerNorm over last dim C (one wave per row)
// ---------------------------------------------------------------------------
__global__ __launch_bounds__(32)
void layernorm_k(const float* __restrict__ x, const float* __restrict__ g,
                 const float* __restrict__ b, float* __restrict__ y, int C)
{
  long row = blockIdx.x;
  const float* xr = x + row * C;
  float* yr = y + row * C;
  int t = threadIdx.x;
  float s = 0.f, s2 = 0.f;
  for (int c = t; c < C; c += 32) { float v = xr[c]; s += v; s2 += v * v; }
  for (int o = 16; o > 0; o >>= 1) { s += __shfl_down(s, o, 32); s2 += __shfl_down(s2, o, 32); }
  s = __shfl(s, 0, 32); s2 = __shfl(s2, 0, 32);
  float mean = s / C;
  float var  = s2 / C - mean * mean;
  float inv  = rsqrtf(var + 1e-5f);
  for (int c = t; c < C; c += 32) yr[c] = (xr[c] - mean) * inv * g[c] + b[c];
}

// ---------------------------------------------------------------------------
// Elementwise / conversion kernels
// ---------------------------------------------------------------------------
__global__ void cvt_bf16_k(const float* __restrict__ x, u16b* __restrict__ y, long n) {
  long i = (long)blockIdx.x * blockDim.x + threadIdx.x;
  if (i < n) y[i] = f2bf(x[i]);
}
// dst[N][M] (bf16) = src[M][N] (f32) — used for weights and OPM activation transposes
__global__ void tcvt_bf16_k(const float* __restrict__ src, u16b* __restrict__ dst, int Mr, int Nc) {
  long i = (long)blockIdx.x * blockDim.x + threadIdx.x;
  long n = (long)Mr * Nc;
  if (i >= n) return;
  long m = i / Nc; long c = i % Nc;
  dst[c * (long)Mr + m] = f2bf(src[i]);
}
__global__ void add_k(float* __restrict__ d, const float* __restrict__ s, long n) {
  long i = (long)blockIdx.x * blockDim.x + threadIdx.x;
  if (i < n) d[i] += s[i];
}
__global__ void mul_k(float* __restrict__ d, const float* __restrict__ a, const float* __restrict__ b, long n) {
  long i = (long)blockIdx.x * blockDim.x + threadIdx.x;
  if (i < n) d[i] = a[i] * b[i];
}
__global__ void muladd_k(float* __restrict__ d, const float* __restrict__ a, const float* __restrict__ b, long n) {
  long i = (long)blockIdx.x * blockDim.x + threadIdx.x;
  if (i < n) d[i] += a[i] * b[i];
}
// per-channel extraction for triangle-mult: dst[c][x][y] = trans ? src[y][x][c] : src[x][y][c]
__global__ void chan_extract_k(const float* __restrict__ src, u16b* __restrict__ dst, int R_, int C, int trans) {
  long i = (long)blockIdx.x * blockDim.x + threadIdx.x;
  long n = (long)R_ * R_ * C;
  if (i >= n) return;
  int c = (int)(i % C);
  long rest = i / C;
  int j = (int)(rest % R_);
  int ii = (int)(rest / R_);
  long o = trans ? ((long)c * R_ * R_ + (long)j * R_ + ii)
                 : ((long)c * R_ * R_ + (long)ii * R_ + j);
  dst[o] = f2bf(src[i]);
}
__global__ void transpose_rr_k(const float* __restrict__ src, float* __restrict__ dst, int R_, int C) {
  long i = (long)blockIdx.x * blockDim.x + threadIdx.x;
  long n = (long)R_ * R_ * C;
  if (i >= n) return;
  int c = (int)(i % C);
  long rest = i / C;
  int j = (int)(rest % R_);
  int ii = (int)(rest / R_);
  dst[((long)j * R_ + ii) * C + c] = src[i];
}
__global__ void addT_k(float* __restrict__ dst, const float* __restrict__ src, int R_, int C) {
  long i = (long)blockIdx.x * blockDim.x + threadIdx.x;
  long n = (long)R_ * R_ * C;
  if (i >= n) return;
  int c = (int)(i % C);
  long rest = i / C;
  int j = (int)(rest % R_);
  int ii = (int)(rest / R_);
  dst[i] += src[((long)j * R_ + ii) * C + c];
}
// small-N dense GEMM (pair bias N=4/8, col k/v N=8), f32 VALU
__global__ void gemm_small_k(const float* __restrict__ A, const float* __restrict__ W,
                             float* __restrict__ C, int N, int K, long M) {
  long i = (long)blockIdx.x * blockDim.x + threadIdx.x;
  if (i >= M * N) return;
  long m = i / N; int n = (int)(i % N);
  const float* a = A + m * K;
  float s = 0.f;
  for (int k = 0; k < K; ++k) s += a[k] * W[(long)k * N + n];
  C[i] = s;
}
// mean over s: y[r*C+c] = (1/S) sum_s x[s*(R*C) + r*C + c]
__global__ void mean_s_k(const float* __restrict__ x, float* __restrict__ y, int R_, int C, int S_) {
  int i = blockIdx.x * blockDim.x + threadIdx.x;
  if (i >= R_ * C) return;
  float s = 0.f;
  long stride = (long)R_ * C;
  for (int ss = 0; ss < S_; ++ss) s += x[(long)ss * stride + i];
  y[i] = s / (float)S_;
}
// col-global gating: y[s,r,c] = g[s,r,c] * oc[r,c]
__global__ void colgate_k(const float* __restrict__ g, const float* __restrict__ oc,
                          float* __restrict__ y, long n, int RC) {
  long i = (long)blockIdx.x * blockDim.x + threadIdx.x;
  if (i < n) y[i] = g[i] * oc[i % RC];
}

// ---------------------------------------------------------------------------
// Fused softmax attention kernels (small head dims -> VALU)
// ---------------------------------------------------------------------------
// MSA row attention: block = (s,h,i); thread j over R keys; D=8
__global__ __launch_bounds__(256)
void msa_row_attn_k(const float* __restrict__ q, const float* __restrict__ k,
                    const float* __restrict__ v, const float* __restrict__ bias,
                    float* __restrict__ o, int R_, int H, int D, float scale)
{
  long bid = blockIdx.x;
  int i = (int)(bid % R_);
  long sh = bid / R_;
  int h = (int)(sh % H);
  long s = sh / H;
  __shared__ float red[256];
  __shared__ float aw[256];
  int j = threadIdx.x;
  int HD = H * D;
  const float* qp = q + (s * R_ + i) * HD + h * D;
  const float* kp = k + (s * R_ + j) * HD + h * D;
  float lg = 0.f;
  for (int d = 0; d < D; ++d) lg += qp[d] * kp[d];
  lg = lg * scale + bias[((long)i * R_ + j) * H + h];
  red[j] = lg; __syncthreads();
  for (int st = 128; st > 0; st >>= 1) { if (j < st) red[j] = fmaxf(red[j], red[j + st]); __syncthreads(); }
  float mx = red[0]; __syncthreads();
  float p = __expf(lg - mx);
  red[j] = p; __syncthreads();
  for (int st = 128; st > 0; st >>= 1) { if (j < st) red[j] += red[j + st]; __syncthreads(); }
  float Z = red[0];
  aw[j] = p / Z; __syncthreads();
  if (j < D) {
    const float* vp = v + (s * R_) * HD + h * D + j;
    float acc = 0.f;
    for (int jj = 0; jj < R_; ++jj) acc += aw[jj] * vp[(long)jj * HD];
    o[(s * R_ + i) * HD + h * D + j] = acc;
  }
}
// MSA column global attention: block = (r,h); keys s over S=512 (2 per thread); D=8
__global__ __launch_bounds__(256)
void msa_col_attn_k(const float* __restrict__ qr, const float* __restrict__ k,
                    const float* __restrict__ v, float* __restrict__ oc,
                    int R_, int S_, int H, int D, float scale)
{
  int h = blockIdx.x % H;
  int r = blockIdx.x / H;
  __shared__ float red[256];
  __shared__ float aw[512];
  int t = threadIdx.x;
  const float* qp = qr + r * H * D + h * D;
  const float* k0 = k + ((long)t * R_ + r) * D;
  const float* k1 = k + ((long)(t + 256) * R_ + r) * D;
  float l0 = 0.f, l1 = 0.f;
  for (int d = 0; d < D; ++d) { l0 += qp[d] * k0[d]; l1 += qp[d] * k1[d]; }
  l0 *= scale; l1 *= scale;
  red[t] = fmaxf(l0, l1); __syncthreads();
  for (int st = 128; st > 0; st >>= 1) { if (t < st) red[t] = fmaxf(red[t], red[t + st]); __syncthreads(); }
  float mx = red[0]; __syncthreads();
  float p0 = __expf(l0 - mx), p1 = __expf(l1 - mx);
  red[t] = p0 + p1; __syncthreads();
  for (int st = 128; st > 0; st >>= 1) { if (t < st) red[t] += red[t + st]; __syncthreads(); }
  float Z = red[0];
  aw[t] = p0 / Z; aw[t + 256] = p1 / Z; __syncthreads();
  if (t < D) {
    const float* vp = v + (long)r * D + t;
    float acc = 0.f;
    long stride = (long)R_ * D;
    for (int s = 0; s < S_; ++s) acc += aw[s] * vp[(long)s * stride];
    oc[r * H * D + h * D + t] = acc;
  }
}
// Triangle attention (starting node): block = (i,j,h); thread t over R keys; D=32
__global__ __launch_bounds__(256)
void tri_attn_k(const float* __restrict__ q, const float* __restrict__ k,
                const float* __restrict__ v, const float* __restrict__ bias,
                float* __restrict__ o, int R_, int H, int D, float scale)
{
  long bid = blockIdx.x;
  int h = (int)(bid % H);
  long ij = bid / H;
  int j = (int)(ij % R_);
  int i = (int)(ij / R_);
  __shared__ float red[256];
  __shared__ float aw[256];
  int t = threadIdx.x;
  int HD = H * D;
  const float* qp = q + ((long)i * R_ + j) * HD + h * D;
  const float* kp = k + ((long)i * R_ + t) * HD + h * D;
  float lg = 0.f;
  for (int d = 0; d < D; ++d) lg += qp[d] * kp[d];
  lg = lg * scale + bias[((long)j * R_ + t) * H + h];
  red[t] = lg; __syncthreads();
  for (int st = 128; st > 0; st >>= 1) { if (t < st) red[t] = fmaxf(red[t], red[t + st]); __syncthreads(); }
  float mx = red[0]; __syncthreads();
  float p = __expf(lg - mx);
  red[t] = p; __syncthreads();
  for (int st = 128; st > 0; st >>= 1) { if (t < st) red[t] += red[t + st]; __syncthreads(); }
  float Z = red[0];
  aw[t] = p / Z; __syncthreads();
  if (t < D) {
    const float* vp = v + ((long)i * R_) * HD + h * D + t;
    float acc = 0.f;
    for (int kk = 0; kk < R_; ++kk) acc += aw[kk] * vp[(long)kk * HD];
    o[((long)i * R_ + j) * HD + h * D + t] = acc;
  }
}

// ---------------------------------------------------------------------------
// Host orchestration
// ---------------------------------------------------------------------------
extern "C" void kernel_launch(void* const* d_in, const int* in_sizes, int n_in,
                              void* d_out, int out_size, void* d_ws, size_t ws_size,
                              hipStream_t stream)
{
  (void)in_sizes; (void)n_in; (void)out_size; (void)ws_size;
  const long S = 512, R = 256, CE = 64, CZ = 128, HM = 8, CM = 8, HP = 4, CTA = 32, CTU = 128, CP = 32;
  const long SR = S * R;   // 131072
  const long RR = R * R;   // 65536

  float* out_m = (float*)d_out;
  float* out_z = out_m + SR * CE;

  // bump allocator in workspace
  char* wp = (char*)d_ws;
  auto alloc = [&](size_t bytes) -> char* { char* p = wp; wp += (bytes + 255) & ~(size_t)255; return p; };
  float* LN  = (float*)alloc((size_t)8388608 * 4);     // LN outputs (max 8.4M f32)
  u16b*  BFA = (u16b*) alloc((size_t)67108864 * 2);    // bf16 activations (max: OPM 67.1M)
  float* Qb  = (float*)alloc((size_t)8388608 * 4);
  float* Kb  = (float*)alloc((size_t)8388608 * 4);
  float* Vb  = (float*)alloc((size_t)8388608 * 4);
  float* Gb  = (float*)alloc((size_t)8388608 * 4);
  float* Ob  = (float*)alloc((size_t)8388608 * 4);
  float* Tb  = (float*)alloc((size_t)33554432 * 4);    // transition hidden / zn / zpT
  float* BIAS= (float*)alloc((size_t)524288 * 4);      // attention pair biases
  u16b*  WT  = (u16b*) alloc((size_t)131072 * 2);      // transposed bf16 weight (max 1024x128)
  u16b*  B1  = (u16b*) alloc((size_t)8388608 * 2);     // chan-extract / OPM A^T
  u16b*  B2  = (u16b*) alloc((size_t)8388608 * 2);
  float* SMq = (float*)alloc((size_t)65536 * 4);       // col-attn mean query
  float* SMo = SMq + 32768;                            // col-attn output [R,64]

  int pi = 2;
  auto P = [&]() { return (const float*)d_in[pi++]; };

  auto LNK  = [&](const float* x, const float* g, const float* b, float* y, long rows, int C) {
    layernorm_k<<<dim3((unsigned)rows), 32, 0, stream>>>(x, g, b, y, C);
  };
  auto CVT  = [&](const float* s, u16b* d, long n) {
    cvt_bf16_k<<<dim3((unsigned)((n + 255) / 256)), 256, 0, stream>>>(s, d, n);
  };
  auto TCVT = [&](const float* s, u16b* d, int Mr, int Nc) {
    long n = (long)Mr * Nc;
    tcvt_bf16_k<<<dim3((unsigned)((n + 255) / 256)), 256, 0, stream>>>(s, d, Mr, Nc);
  };
  auto GEMM = [&](const u16b* A, long lda, long sA, const u16b* Bt, long ldb, long sB,
                  void* C, long ldcM, long ldcN, long sC, const float* bias, float scale,
                  long M, long N, long K, int batch, int act, int opst) {
    int tilesN = (int)(N / 32);
    int tilesTotal = (int)((M / 32) * (N / 32));
    dim3 g((unsigned)((tilesTotal + 7) / 8), (unsigned)batch);
    wmma_gemm_bf16_k<<<g, 256, 0, stream>>>((const __bf16*)A, lda, sA, (const __bf16*)Bt, ldb, sB,
                                            C, ldcM, ldcN, sC, bias, scale,
                                            (int)N, (int)K, tilesN, tilesTotal, act, opst);
  };
  auto GEMMW = [&](const u16b* Abf, const float* Wf, long K, long N, const float* bias,
                   float* C, long M, int act) {
    TCVT(Wf, WT, (int)K, (int)N);
    GEMM(Abf, K, 0, WT, K, 0, C, N, 1, 0, bias, 1.0f, M, N, K, 1, act, 0);
  };
  auto ADD    = [&](float* d, const float* s, long n) { add_k<<<dim3((unsigned)((n + 255) / 256)), 256, 0, stream>>>(d, s, n); };
  auto MUL    = [&](float* d, const float* a, const float* b, long n) { mul_k<<<dim3((unsigned)((n + 255) / 256)), 256, 0, stream>>>(d, a, b, n); };
  auto MULADD = [&](float* d, const float* a, const float* b, long n) { muladd_k<<<dim3((unsigned)((n + 255) / 256)), 256, 0, stream>>>(d, a, b, n); };
  auto SMALLG = [&](const float* A, const float* W, float* C, long M, int N, int K) {
    long n = M * N;
    gemm_small_k<<<dim3((unsigned)((n + 255) / 256)), 256, 0, stream>>>(A, W, C, N, K, M);
  };

  // ---- init residual state ----
  hipMemcpyAsync(out_m, d_in[0], (size_t)SR * CE * 4, hipMemcpyDeviceToDevice, stream);
  hipMemcpyAsync(out_z, d_in[1], (size_t)RR * CZ * 4, hipMemcpyDeviceToDevice, stream);

  // ---- 1. MSA row attention with pair bias ----
  {
    const float* ln_m_g = P(); const float* ln_m_b = P();
    const float* ln_z_g = P(); const float* ln_z_b = P();
    const float* wq = P(); const float* wk = P(); const float* wv = P(); const float* wb = P();
    const float* wg = P(); const float* bg = P(); const float* wo = P(); const float* bo = P();
    LNK(out_m, ln_m_g, ln_m_b, LN, SR, 64);
    CVT(LN, BFA, SR * 64);
    LNK(out_z, ln_z_g, ln_z_b, Tb, RR, 128);                 // zn
    GEMMW(BFA, wq, 64, 64, nullptr, Qb, SR, 0);
    GEMMW(BFA, wk, 64, 64, nullptr, Kb, SR, 0);
    GEMMW(BFA, wv, 64, 64, nullptr, Vb, SR, 0);
    GEMMW(BFA, wg, 64, 64, bg, Gb, SR, 2);                   // sigmoid gate
    SMALLG(Tb, wb, BIAS, RR, 8, 128);                        // pair bias [R*R,8]
    msa_row_attn_k<<<dim3((unsigned)(S * HM * R)), 256, 0, stream>>>(Qb, Kb, Vb, BIAS, Ob,
                                                                     (int)R, (int)HM, (int)CM, 0.3535533905932738f);
    MUL(Ob, Gb, Ob, SR * 64);
    CVT(Ob, BFA, SR * 64);
    GEMMW(BFA, wo, 64, 64, bo, Qb, SR, 0);
    ADD(out_m, Qb, SR * 64);
  }

  // ---- 2. MSA column global attention ----
  {
    const float* lg = P(); const float* lb = P();
    const float* wq = P(); const float* wk = P(); const float* wv = P();
    const float* wg = P(); const float* bg = P(); const float* wo = P(); const float* bo = P();
    LNK(out_m, lg, lb, LN, SR, 64);
    CVT(LN, BFA, SR * 64);
    GEMMW(BFA, wq, 64, 64, nullptr, Qb, SR, 0);
    mean_s_k<<<dim3((unsigned)((R * 64 + 255) / 256)), 256, 0, stream>>>(Qb, SMq, (int)R, 64, (int)S);
    SMALLG(LN, wk, Kb, SR, 8, 64);
    SMALLG(LN, wv, Vb, SR, 8, 64);
    GEMMW(BFA, wg, 64, 64, bg, Gb, SR, 2);
    msa_col_attn_k<<<dim3((unsigned)(R * HM)), 256, 0, stream>>>(SMq, Kb, Vb, SMo,
                                                                 (int)R, (int)S, (int)HM, (int)CM, 0.3535533905932738f);
    colgate_k<<<dim3((unsigned)((SR * 64 + 255) / 256)), 256, 0, stream>>>(Gb, SMo, Ob, SR * 64, (int)(R * 64));
    CVT(Ob, BFA, SR * 64);
    GEMMW(BFA, wo, 64, 64, bo, Qb, SR, 0);
    ADD(out_m, Qb, SR * 64);
  }

  // ---- 3. MSA transition (64 -> 256 -> 64) ----
  {
    const float* lg = P(); const float* lb = P();
    const float* w1 = P(); const float* b1 = P(); const float* w2 = P(); const float* b2 = P();
    LNK(out_m, lg, lb, LN, SR, 64);
    CVT(LN, BFA, SR * 64);
    GEMMW(BFA, w1, 64, 256, b1, Tb, SR, 1);                  // relu
    CVT(Tb, BFA, SR * 256);
    GEMMW(BFA, w2, 256, 64, b2, Qb, SR, 0);
    ADD(out_m, Qb, SR * 64);
  }

  // ---- 4. Outer product mean (the big one: 8192x8192x512 WMMA GEMM) ----
  {
    const float* lg = P(); const float* lb = P();
    const float* wa = P(); const float* ba = P(); const float* wb2 = P(); const float* bb = P();
    const float* wo = P(); const float* bo = P();
    LNK(out_m, lg, lb, LN, SR, 64);
    CVT(LN, BFA, SR * 64);
    GEMMW(BFA, wa,  64, 32, ba, Qb, SR, 0);                  // a [S, R*32]
    GEMMW(BFA, wb2, 64, 32, bb, Kb, SR, 0);                  // b [S, R*32]
    TCVT(Qb, B1, (int)S, (int)(R * CP));                     // A^T bf16 [8192,512]
    TCVT(Kb, B2, (int)S, (int)(R * CP));
    // o[(i,a)][(j,b)] stored directly as bf16 in (i,j,a*32+b) order into BFA
    GEMM(B1, S, 0, B2, S, 0, BFA, 0, 0, 0, nullptr, 1.0f / (float)S,
         R * CP, R * CP, S, 1, 0, /*opst=*/1);
    GEMMW(BFA, wo, 1024, 128, bo, Qb, RR, 0);
    ADD(out_z, Qb, RR * 128);
  }

  // ---- 5./6. Triangle multiplicative update (outgoing, incoming) ----
  auto TRIMULT = [&](bool outgoing) {
    const float* ln_g = P(); const float* ln_b = P();
    const float* wa = P(); const float* ba = P(); const float* wag = P(); const float* bag = P();
    const float* wb = P(); const float* bb = P(); const float* wbg = P(); const float* bbg = P();
    const float* wg = P(); const float* bg = P(); const float* lno_g = P(); const float* lno_b = P();
    const float* wo = P(); const float* bo = P();
    LNK(out_z, ln_g, ln_b, LN, RR, 128);
    CVT(LN, BFA, RR * 128);
    GEMMW(BFA, wa,  128, 128, ba,  Qb, RR, 0);
    GEMMW(BFA, wag, 128, 128, bag, Kb, RR, 2);
    MUL(Qb, Kb, Qb, RR * 128);                               // a = sig(gate)*lin
    GEMMW(BFA, wb,  128, 128, bb,  Kb, RR, 0);
    GEMMW(BFA, wbg, 128, 128, bbg, Vb, RR, 2);
    MUL(Kb, Vb, Kb, RR * 128);                               // b
    GEMMW(BFA, wg,  128, 128, bg,  Gb, RR, 2);               // g
    int tr = outgoing ? 0 : 1;
    chan_extract_k<<<dim3((unsigned)((RR * 128 + 255) / 256)), 256, 0, stream>>>(Qb, B1, (int)R, 128, tr);
    chan_extract_k<<<dim3((unsigned)((RR * 128 + 255) / 256)), 256, 0, stream>>>(Kb, B2, (int)R, 128, tr);
    // 128 batched [256x256x256] GEMMs; store x in [i,j,c] layout
    GEMM(B1, R, RR, B2, R, RR, Ob, R * CTU, CTU, 1, nullptr, 1.0f, R, R, R, 128, 0, 0);
    LNK(Ob, lno_g, lno_b, LN, RR, 128);
    CVT(LN, BFA, RR * 128);
    GEMMW(BFA, wo, 128, 128, bo, Qb, RR, 0);
    MULADD(out_z, Gb, Qb, RR * 128);                         // zp += g * proj
  };
  TRIMULT(true);
  TRIMULT(false);

  // ---- 7./8. Triangle attention (starting / ending node) ----
  auto TRIATTN = [&](const float* zin, float* yout) {
    const float* ln_g = P(); const float* ln_b = P();
    const float* wq = P(); const float* wk = P(); const float* wv = P(); const float* wb = P();
    const float* wg = P(); const float* bg = P(); const float* wo = P(); const float* bo = P();
    LNK(zin, ln_g, ln_b, LN, RR, 128);
    CVT(LN, BFA, RR * 128);
    GEMMW(BFA, wq, 128, 128, nullptr, Qb, RR, 0);
    GEMMW(BFA, wk, 128, 128, nullptr, Kb, RR, 0);
    GEMMW(BFA, wv, 128, 128, nullptr, Vb, RR, 0);
    GEMMW(BFA, wg, 128, 128, bg, Gb, RR, 2);
    SMALLG(LN, wb, BIAS, RR, 4, 128);                        // bias [R*R,4]
    tri_attn_k<<<dim3((unsigned)(HP * RR)), 256, 0, stream>>>(Qb, Kb, Vb, BIAS, Ob,
                                                              (int)R, (int)HP, (int)CTA, 0.1767766952966369f);
    MUL(Ob, Gb, Ob, RR * 128);
    CVT(Ob, BFA, RR * 128);
    GEMMW(BFA, wo, 128, 128, bo, yout, RR, 0);
  };
  TRIATTN(out_z, Qb);
  ADD(out_z, Qb, RR * 128);
  transpose_rr_k<<<dim3((unsigned)((RR * 128 + 255) / 256)), 256, 0, stream>>>(out_z, Tb, (int)R, 128);
  TRIATTN(Tb, Qb);
  addT_k<<<dim3((unsigned)((RR * 128 + 255) / 256)), 256, 0, stream>>>(out_z, Qb, (int)R, 128);

  // ---- 9. Pair transition (128 -> 512 -> 128) ----
  {
    const float* lg = P(); const float* lb = P();
    const float* w1 = P(); const float* b1 = P(); const float* w2 = P(); const float* b2 = P();
    LNK(out_z, lg, lb, LN, RR, 128);
    CVT(LN, BFA, RR * 128);
    GEMMW(BFA, w1, 128, 512, b1, Tb, RR, 1);                 // relu
    CVT(Tb, BFA, RR * 512);
    GEMMW(BFA, w2, 512, 128, b2, Qb, RR, 0);
    ADD(out_z, Qb, RR * 128);
  }
}
